// HGTMessage_55705725829588
// MI455X (gfx1250) — matrix operand order repaired and checked by reading, not compiled, not run on hardware
//
#include <hip/hip_runtime.h>

// ---------------- problem constants ----------------
#define E_N     131072
#define IN_SZ   128
#define NH      8
#define DK      16
#define NTY     4      // node types
#define NET     8      // edge types
#define MAXLEN  240

// ---------------- tiling ----------------
#define WAVES            4           // waves per block (128 threads)
#define ETILES           2           // 16-edge tiles per wave (M blocking)
#define EDGES_PER_WAVE   (16 * ETILES)
#define EDGES_PER_BLOCK  (WAVES * EDGES_PER_WAVE)

#define SA_HALFS_PER_WAVE (ETILES * 16 * 136)             // padded h_hat tile
#define SA_BYTES          (WAVES * SA_HALFS_PER_WAVE * 2) // 34816
#define W1_PART_HALFS     (16 * 32 * 16)                  // 8192 halfs = 16KB per nt
#define W1_PART_BYTES     16384
// smem region: [ buf0 16KB | buf1 16KB | sW2 16KB ]; sA (34816B) unions at 0
#define SMEM_BYTES        49152

typedef _Float16 v16h __attribute__((ext_vector_type(16)));
typedef _Float16 v8h  __attribute__((ext_vector_type(8)));
typedef _Float16 v4h  __attribute__((ext_vector_type(4)));
typedef float    v8f  __attribute__((ext_vector_type(8)));
typedef float    v4f  __attribute__((ext_vector_type(4)));

// ---------------------------------------------------------------------------
// Prep A: positional table P[240][128] = emb @ lin_W^T + lin_b   (f32)
// ---------------------------------------------------------------------------
__global__ void pos_table_kernel(const float* __restrict__ emb,
                                 const float* __restrict__ linW,
                                 const float* __restrict__ linb,
                                 float* __restrict__ P) {
    int r = blockIdx.x;          // 0..239
    int c = threadIdx.x;         // 0..127
    const float* er = emb  + r * IN_SZ;
    const float* wr = linW + c * IN_SZ;    // (emb @ W^T)[r,c] = sum_i emb[r,i]*W[c,i]
    float s = linb[c];
#pragma unroll 8
    for (int i = 0; i < IN_SZ; ++i) s += er[i] * wr[i];
    P[r * IN_SZ + c] = s;
}

// ---------------------------------------------------------------------------
// Prep B: pack K_W / V_W into f16 WMMA-B-operand layout, nt-major so each
// nt's 16KB region is contiguous for async LDS staging.
// Layout: W1[((nt*16 + t*4 + c)*32 + lane)*16 + j]
//   nt = 2*h + part (part 0 = K head h, part 1 = V head h)
//   B element (k, n): n = lane%16, k = c*32 + (j%8) + 8*(lane/16) + 16*(j/8)
// ---------------------------------------------------------------------------
__global__ void pack_w1_kernel(const float* __restrict__ KW,
                               const float* __restrict__ VW,
                               _Float16* __restrict__ W1) {
    int idx  = blockIdx.x * 256 + threadIdx.x;   // 131072 total
    int j    = idx & 15;
    int lane = (idx >> 4) & 31;
    int c    = (idx >> 9) & 3;
    int t    = (idx >> 11) & 3;
    int nt   = (idx >> 13) & 15;
    int n    = lane & 15;
    int k    = c * 32 + (j & 7) + 8 * (lane >> 4) + 16 * (j >> 3);
    int part = nt & 1, h = nt >> 1;
    int o    = h * 16 + n;
    const float* src = part ? VW : KW;
    W1[idx] = (_Float16)src[(t * IN_SZ + o) * IN_SZ + k];
}

// ---------------------------------------------------------------------------
// Prep C: pack stage-2 B tiles (32x16, f16) per etype:
//   s=0: Batt[k][o] = (k<16)  ? Watt[et][o][k]     : 0
//   s=1: Bmsg[k][o] = (k>=16) ? Wmsg[et][o][k-16]  : 0
// ---------------------------------------------------------------------------
__global__ void pack_w2_kernel(const float* __restrict__ Watt,
                               const float* __restrict__ Wmsg,
                               _Float16* __restrict__ W2) {
    int idx  = blockIdx.x * 256 + threadIdx.x;   // 8192 total
    int j    = idx & 15;
    int lane = (idx >> 4) & 31;
    int s    = (idx >> 9) & 1;
    int et   = (idx >> 10) & 7;
    int o    = lane & 15;
    int k    = (j & 7) + 8 * (lane >> 4) + 16 * (j >> 3);
    float v  = 0.0f;
    if (s == 0) { if (k < 16)  v = Watt[(et * 16 + o) * 16 + k]; }
    else        { if (k >= 16) v = Wmsg[(et * 16 + o) * 16 + (k - 16)]; }
    W2[idx] = (_Float16)v;
}

// async copy of one 16KB W1 part region (global -> LDS), 128 threads x 8 x B128
__device__ __forceinline__ void async_stage_w1(const _Float16* W1g, int nt,
                                               unsigned lds_off, int tid) {
    unsigned long long gbase = (unsigned long long)(const void*)(W1g + nt * W1_PART_HALFS);
#pragma unroll
    for (int it = 0; it < 8; ++it) {
        unsigned goff = (unsigned)(tid * 16 + it * 2048);
        unsigned loff = lds_off + goff;
        asm volatile("global_load_async_to_lds_b128 %0, %1, %2"
                     :: "v"(loff), "v"(goff), "s"(gbase) : "memory");
    }
}

// ---------------------------------------------------------------------------
// Main kernel.
// ---------------------------------------------------------------------------
__global__ __launch_bounds__(32 * WAVES)
void hgt_main_kernel(const int*   __restrict__ etype,
                     const int*   __restrict__ tau_s,
                     const float* __restrict__ h_s,
                     const float* __restrict__ Q_t,
                     const int*   __restrict__ dt,
                     const float* __restrict__ Kb,
                     const float* __restrict__ Vb,
                     const float* __restrict__ mu,
                     const float* __restrict__ P,
                     const _Float16* __restrict__ W1g,
                     const _Float16* __restrict__ W2g,
                     float* __restrict__ out_att,
                     float* __restrict__ out_M) {
    __shared__ __align__(32) char smem[SMEM_BYTES];   // sA, then buf0|buf1|sW2
    __shared__ _Float16 sKV[WAVES][ETILES][16][48];   // 96B rows, 16B aligned
    __shared__ int sTau[WAVES][EDGES_PER_WAVE];
    __shared__ int sEt [WAVES][EDGES_PER_WAVE];
    __shared__ int sDt [WAVES][EDGES_PER_WAVE];

    _Float16* sA  = (_Float16*)smem;
    _Float16* sW2 = (_Float16*)(smem + 32768);

    const int tid  = threadIdx.x;
    const int lane = tid & 31;
    const int w    = tid >> 5;
    const long long e0 = (long long)(blockIdx.x * WAVES + w) * EDGES_PER_WAVE;

    // ---- phase 0: per-edge metadata (one lane per edge) ----
    {
        long long e = e0 + lane;
        sTau[w][lane] = tau_s[e];
        sEt [w][lane] = etype[e];
        sDt [w][lane] = dt[e];
    }
    __syncthreads();

    // ---- phase 1: h_hat = h_s + P[dt] -> f16 in sA ----
    for (int u = 0; u < ETILES; ++u) {
        for (int i = 0; i < 16; ++i) {
            long long e = e0 + u * 16 + i;
            int d = sDt[w][u * 16 + i];
            v4f hv = *(const v4f*)(h_s + e * IN_SZ + lane * 4);
            v4f pv = *(const v4f*)(P + (long long)d * IN_SZ + lane * 4);
            v4h o;
#pragma unroll
            for (int q = 0; q < 4; ++q) o[q] = (_Float16)(hv[q] + pv[q]);
            *(v4h*)&sA[(w * ETILES + u) * 16 * 136 + i * 136 + lane * 4] = o;
        }
    }
    __syncthreads();

    const int mrow   = lane & 15;     // A row (edge within tile) / D column
    const int hiHalf = lane >> 4;

    // ---- load A operand chunks per ISA 16-bit 16x32 A layout ----
    v16h A[ETILES][4];
#pragma unroll
    for (int u = 0; u < ETILES; ++u) {
#pragma unroll
        for (int c = 0; c < 4; ++c) {
            const _Float16* base =
                &sA[(w * ETILES + u) * 16 * 136 + mrow * 136 + c * 32 + hiHalf * 8];
            v8h lo = *(const v8h*)base;
            v8h hi = *(const v8h*)(base + 16);
            v16h a;
#pragma unroll
            for (int q = 0; q < 8; ++q) { a[q] = lo[q]; a[8 + q] = hi[q]; }
            A[u][c] = a;
        }
    }
    int tauM[ETILES], etM[ETILES];
#pragma unroll
    for (int u = 0; u < ETILES; ++u) {
        tauM[u] = sTau[w][u * 16 + mrow];
        etM [u] = sEt [w][u * 16 + mrow];
    }
    __syncthreads();   // sA fully consumed; smem region may now be re-staged

    const unsigned smem_base = (unsigned)(uintptr_t)(void*)smem;  // LDS offset

    // ---- stage W2 once (sync) + async-stage buf0 with nt=0 ----
#pragma unroll
    for (int it = 0; it < 4; ++it)
        ((v16h*)sW2)[tid + it * 128] = ((const v16h*)W2g)[tid + it * 128];
    async_stage_w1(W1g, 0, smem_base, tid);
    asm volatile("s_wait_asynccnt 0x0" ::: "memory");
    __syncthreads();

    for (int h = 0; h < NH; ++h) {
        // ================= stage 1: K (part 0) and V (part 1) tiles =========
        for (int part = 0; part < 2; ++part) {
            const int nt  = h * 2 + part;
            const int cur = nt & 1;
            // kick off async copy of next part into the other buffer
            if (nt + 1 < 16)
                async_stage_w1(W1g, nt + 1, smem_base + (cur ^ 1) * W1_PART_BYTES, tid);
            const _Float16* sW1c = (const _Float16*)(smem + cur * W1_PART_BYTES);

            const float* bias = part ? Vb : Kb;
            v8f acc[ETILES];
#pragma unroll
            for (int u = 0; u < ETILES; ++u) {
#pragma unroll
                for (int v = 0; v < 8; ++v) {
                    int m = v + 8 * hiHalf;
                    int tv = sTau[w][u * 16 + m];
                    acc[u][v] = bias[tv * IN_SZ + h * 16 + mrow];
                }
            }
#pragma unroll
            for (int t = 0; t < NTY; ++t) {
                _Float16 msk[ETILES];
#pragma unroll
                for (int u = 0; u < ETILES; ++u) {
                    int tm = tauM[u];
                    asm volatile("" : "+v"(tm));   // block hoisting of masked A
                    msk[u] = (tm == t) ? (_Float16)1.0f : (_Float16)0.0f;
                }
#pragma unroll
                for (int c = 0; c < 4; ++c) {
                    v16h b = *(const v16h*)(sW1c + (((t * 4 + c) * 32 + lane) << 4));
#pragma unroll
                    for (int u = 0; u < ETILES; ++u) {
                        v16h am = A[u][c] * msk[u];   // zero rows of other types
                        acc[u] = __builtin_amdgcn_wmma_f32_16x16x32_f16(
                            false, am, false, b, (short)0, acc[u], false, false);
                    }
                }
            }
            // stash D tile -> LDS as [edge][k] halfs (K at k 0..15, V at 16..31)
#pragma unroll
            for (int u = 0; u < ETILES; ++u) {
#pragma unroll
                for (int v = 0; v < 8; ++v) {
                    int m = v + 8 * hiHalf;
                    sKV[w][u][m][part * 16 + mrow] = (_Float16)acc[u][v];
                }
            }
            asm volatile("s_wait_asynccnt 0x0" ::: "memory");
            __syncthreads();   // buffers swapped; next part may overwrite buf[cur]
        }

        // ================= stage 2: att_k and M for this head ===============
        // (sKV is strictly per-wave: no extra barrier needed)
        v16h A2[ETILES];
#pragma unroll
        for (int u = 0; u < ETILES; ++u) {
            const _Float16* base = &sKV[w][u][mrow][hiHalf * 8];
            v8h lo = *(const v8h*)base;
            v8h hi = *(const v8h*)(base + 16);
            v16h a;
#pragma unroll
            for (int q = 0; q < 8; ++q) { a[q] = lo[q]; a[8 + q] = hi[q]; }
            A2[u] = a;
        }
        v8f aAtt[ETILES], aMsg[ETILES];
#pragma unroll
        for (int u = 0; u < ETILES; ++u)
#pragma unroll
            for (int q = 0; q < 8; ++q) { aAtt[u][q] = 0.0f; aMsg[u][q] = 0.0f; }

#pragma unroll
        for (int et = 0; et < NET; ++et) {
            v16h batt = *(const v16h*)(sW2 + (((et * 2 + 0) * 32 + lane) << 4));
            v16h bmsg = *(const v16h*)(sW2 + (((et * 2 + 1) * 32 + lane) << 4));
#pragma unroll
            for (int u = 0; u < ETILES; ++u) {
                int em = etM[u];
                asm volatile("" : "+v"(em));
                _Float16 msk = (em == et) ? (_Float16)1.0f : (_Float16)0.0f;
                v16h am = A2[u] * msk;
                aAtt[u] = __builtin_amdgcn_wmma_f32_16x16x32_f16(
                    false, am, false, batt, (short)0, aAtt[u], false, false);
                aMsg[u] = __builtin_amdgcn_wmma_f32_16x16x32_f16(
                    false, am, false, bmsg, (short)0, aMsg[u], false, false);
            }
        }

        // ---- write M, reduce att over o (lanes within 16-group) ----
#pragma unroll
        for (int u = 0; u < ETILES; ++u) {
            float p[8];
#pragma unroll
            for (int v = 0; v < 8; ++v) {
                int m = v + 8 * hiHalf;
                long long e = e0 + u * 16 + m;
                out_M[e * (NH * DK) + h * DK + mrow] = aMsg[u][v];
                float q = Q_t[e * (NH * DK) + h * DK + mrow];
                p[v] = aAtt[u][v] * q;
            }
#pragma unroll
            for (int off = 1; off < 16; off <<= 1)
#pragma unroll
                for (int v = 0; v < 8; ++v) p[v] += __shfl_xor(p[v], off, 16);
            if (mrow == 0) {
#pragma unroll
                for (int v = 0; v < 8; ++v) {
                    int m = v + 8 * hiHalf;
                    long long e = e0 + u * 16 + m;
                    int ev = sEt[w][u * 16 + m];
                    out_att[e * NH + h] = p[v] * mu[ev * NH + h] * 0.25f;  // 1/sqrt(16)
                }
            }
        }
    }
}

// ---------------------------------------------------------------------------
extern "C" void kernel_launch(void* const* d_in, const int* in_sizes, int n_in,
                              void* d_out, int out_size, void* d_ws, size_t ws_size,
                              hipStream_t stream) {
    const int*   etype = (const int*)  d_in[0];
    const int*   tau_s = (const int*)  d_in[1];
    const float* h_s   = (const float*)d_in[2];
    const float* Q_t   = (const float*)d_in[3];
    const int*   dtv   = (const int*)  d_in[4];
    const float* emb   = (const float*)d_in[5];
    const float* linW  = (const float*)d_in[6];
    const float* linb  = (const float*)d_in[7];
    const float* K_W   = (const float*)d_in[8];
    const float* K_b   = (const float*)d_in[9];
    const float* V_W   = (const float*)d_in[10];
    const float* V_b   = (const float*)d_in[11];
    const float* WattW = (const float*)d_in[12];
    const float* WmsgW = (const float*)d_in[13];
    const float* mu    = (const float*)d_in[14];

    // workspace layout: P (122880B, padded to 128KB) | W1 (256KB) | W2 (16KB)
    float*    P  = (float*)d_ws;
    _Float16* W1 = (_Float16*)((char*)d_ws + 131072);
    _Float16* W2 = (_Float16*)((char*)d_ws + 131072 + 262144);

    float* out_att = (float*)d_out;
    float* out_M   = out_att + (size_t)E_N * NH;

    pos_table_kernel<<<MAXLEN, IN_SZ, 0, stream>>>(emb, linW, linb, P);
    pack_w1_kernel<<<512, 256, 0, stream>>>(K_W, V_W, W1);
    pack_w2_kernel<<<32, 256, 0, stream>>>(WattW, WmsgW, W2);
    hgt_main_kernel<<<E_N / EDGES_PER_BLOCK, 32 * WAVES, 0, stream>>>(
        etype, tau_s, h_s, Q_t, dtv, K_b, V_b, mu, P, W1, W2, out_att, out_M);
}
